// MLP_adap_2_layer_1_adp_6227702579415
// MI455X (gfx1250) — compile-verified, matrix operand order checked by reading, not run on hardware
//
#include <hip/hip_runtime.h>
#include <hip/hip_bf16.h>

// MI455X / gfx1250 implementation.
// Compute-bound GEMM pipeline -> v_wmma_f32_16x16x32_f16 everywhere.
// Adapter einsums re-associated to avoid materializing [B,512,32] tensors.

#define B_   4096
#define F_   30
#define E_   16
#define IN_  480
#define H1_  1024
#define H2_  512
#define K_   32
#define HH_  512
#define D_   3
#define V_   10000
#define EPS_ 1e-5f

typedef _Float16 f16;
typedef __attribute__((ext_vector_type(8)))  f16   v8h;
typedef __attribute__((ext_vector_type(16))) f16   v16h;
typedef __attribute__((ext_vector_type(8)))  float v8f;

// ---------------------------------------------------------------- gather ----
// emb[b, f*16+e] = tables[f, x[f,b], e], converted to f16 for WMMA A-matrix.
__global__ __launch_bounds__(256)
void gather_embed(const int* __restrict__ x, const float* __restrict__ tables,
                  f16* __restrict__ emb_h) {
  int idx = blockIdx.x * 256 + threadIdx.x;           // B_*F_ threads
  if (idx >= B_ * F_) return;
  int f = idx / B_;
  int b = idx - f * B_;
  const float* src = tables + ((size_t)f * V_ + x[f * B_ + b]) * E_;
  f16* dst = emb_h + (size_t)b * IN_ + f * E_;
#pragma unroll
  for (int e = 0; e < E_; ++e) dst[e] = (f16)src[e];
}

// ------------------------------------------------- weight transpose+cvt ----
// W[Kd,N] f32 (row-major) -> Wt[N,Kd] f16 so the WMMA B-fragment loads are
// contiguous per output column.
__global__ __launch_bounds__(256)
void convT(const float* __restrict__ W, f16* __restrict__ Wt, int Kd, int N) {
  int idx = blockIdx.x * 256 + threadIdx.x;
  if (idx >= Kd * N) return;
  int k = idx / N;
  int n = idx - k * N;
  Wt[(size_t)n * Kd + k] = (f16)W[(size_t)k * N + n];
}

// ------------------------------------------------------------ WMMA GEMM ----
// Y[B_,N] = A[B_,Kd](f16) @ Wt[N,Kd]^T(f16)  (+ bias)  (+ column stats)
// 8 waves/block, 4x2 wave grid -> 64x32 tile/block; 16x16 per wave.
// flags: bit0 = add bias, bit1 = atomically accumulate colsum / colsumsq.
__global__ __launch_bounds__(256)
void wmma_gemm(const f16* __restrict__ A, const f16* __restrict__ Wt,
               const float* __restrict__ bias, float* __restrict__ Y,
               float* __restrict__ stats, int Kd, int N, int flags) {
  const int lane = threadIdx.x & 31;
  const int wid  = threadIdx.x >> 5;
  const int m0   = blockIdx.x * 64 + (wid >> 1) * 16;
  const int n0   = blockIdx.y * 32 + (wid & 1) * 16;
  const int half = lane >> 4;     // lane half selects K sub-ranges (ISA 7.12.2)
  const int l15  = lane & 15;

  const f16* ap = A  + (size_t)(m0 + l15) * Kd + half * 8;  // A row, 16B aligned
  const f16* bp = Wt + (size_t)(n0 + l15) * Kd + half * 8;  // W^T row (= column)

  v8f c = {};
  for (int k0 = 0; k0 < Kd; k0 += 32) {
    v8h alo = *(const v8h*)(ap + k0);
    v8h ahi = *(const v8h*)(ap + k0 + 16);
    v8h blo = *(const v8h*)(bp + k0);
    v8h bhi = *(const v8h*)(bp + k0 + 16);
    v16h av = __builtin_shufflevector(alo, ahi, 0,1,2,3,4,5,6,7,8,9,10,11,12,13,14,15);
    v16h bv = __builtin_shufflevector(blo, bhi, 0,1,2,3,4,5,6,7,8,9,10,11,12,13,14,15);
    c = __builtin_amdgcn_wmma_f32_16x16x32_f16(false, av, false, bv,
                                               (short)0, c, false, false);
  }

  // C layout: lane holds column n = n0 + (lane&15), rows m0 + 8*(lane>>4) + r.
  const int n  = n0 + l15;
  const int mb = m0 + half * 8;
  const float bs = (flags & 1) ? bias[n] : 0.f;
  float s = 0.f, sq = 0.f;
#pragma unroll
  for (int r = 0; r < 8; ++r) {
    float v = c[r] + bs;
    Y[(size_t)(mb + r) * N + n] = v;
    s += v; sq += v * v;
  }
  if (flags & 2) {
    atomicAdd(&stats[n], s);            // column sum
    atomicAdd(&stats[H1_ + n], sq);     // column sum of squares
  }
}

// ------------------------------------------------------------- BN + ReLU ----
// Biased batch variance (jnp.var default). Writes f32 and/or f16 copies.
__global__ __launch_bounds__(256)
void bn_relu(const float* __restrict__ Y, const float* __restrict__ stats,
             const float* __restrict__ g, const float* __restrict__ bt,
             float* __restrict__ outF, f16* __restrict__ outH, int N) {
  int idx = blockIdx.x * 256 + threadIdx.x;
  if (idx >= B_ * N) return;
  int n = idx % N;
  float mean = stats[n] * (1.f / B_);
  float var  = stats[H1_ + n] * (1.f / B_) - mean * mean;
  float v = (Y[idx] - mean) * rsqrtf(var + EPS_) * g[n] + bt[n];
  v = v > 0.f ? v : 0.f;
  if (outF) outF[idx] = v;
  if (outH) outH[idx] = (f16)v;
}

// ---------------------------------------------- adapter BN (ddof=1) + res --
__global__ __launch_bounds__(256)
void bn2_residual(const float* __restrict__ Y, const float* __restrict__ stats,
                  const float* __restrict__ g, const float* __restrict__ bt,
                  float* __restrict__ di /* in-out */) {
  int idx = blockIdx.x * 256 + threadIdx.x;
  if (idx >= B_ * H2_) return;
  int n = idx % H2_;
  float mean = stats[n] * (1.f / B_);
  float var  = (stats[H1_ + n] - (float)B_ * mean * mean) * (1.f / (B_ - 1));
  float v = (Y[idx] - mean) * rsqrtf(var + EPS_) * g[n] + bt[n];
  di[idx] += v;
}

// ----------------------------------------------------- per-sample adapter --
// One wave per batch row (8 rows / 256-thread block). Computes:
//   u = p @ hyper[b]; t = sigmoid(u @ v0 + bl0); r = t @ u1; m = r @ hyper[b]
// writing m as f16 for the final WMMA GEMM against v1.
__global__ __launch_bounds__(256)
void adapter_mid(const float* __restrict__ p, const float* __restrict__ hyper,
                 const float* __restrict__ v0, const float* __restrict__ u1,
                 const float* __restrict__ bl0, f16* __restrict__ m_h) {
  __shared__ float pS[8][K_], uS[8][K_], sS[8][K_], rS[8][K_];
  int w = threadIdx.x >> 5;
  int j = threadIdx.x & 31;
  int b = blockIdx.x * 8 + w;
  const float* hb = hyper + (size_t)b * (K_ * K_);

  pS[w][j] = p[(size_t)b * K_ + j];
  __syncthreads();
  float u = 0.f;
#pragma unroll
  for (int i = 0; i < K_; ++i) u += pS[w][i] * hb[i * K_ + j];
  uS[w][j] = u;
  __syncthreads();
  float t = bl0[j];
#pragma unroll
  for (int k = 0; k < K_; ++k) t += uS[w][k] * v0[k * K_ + j];
  float s = 1.f / (1.f + expf(-t));
  sS[w][j] = s;
  __syncthreads();
  float r = 0.f;
#pragma unroll
  for (int i = 0; i < K_; ++i) r += sS[w][i] * u1[i * K_ + j];
  rS[w][j] = r;
  __syncthreads();
  float m = 0.f;
#pragma unroll
  for (int i = 0; i < K_; ++i) m += rS[w][i] * hb[i * K_ + j];
  m_h[(size_t)b * K_ + j] = (f16)m;
}

// ----------------------------------------------------------- final head ----
// One wave per row: sigmoid(di2 . dWo[d] + dbo[d]); write only where
// domain_id == d. Every row's domain is in [0,D), so d_out fully covered.
__global__ __launch_bounds__(256)
void final_head(const float* __restrict__ di2, const float* __restrict__ Wo,
                const float* __restrict__ bo, const int* __restrict__ x,
                int d, float* __restrict__ out) {
  int w = threadIdx.x >> 5;
  int lane = threadIdx.x & 31;
  int b = blockIdx.x * 8 + w;
  const float* row = di2 + (size_t)b * H2_;
  float acc = 0.f;
  for (int i = lane; i < H2_; i += 32) acc += row[i] * Wo[i];
#pragma unroll
  for (int off = 16; off; off >>= 1) acc += __shfl_xor(acc, off, 32);
  if (lane == 0 && x[(F_ - 1) * B_ + b] == d)
    out[b] = 1.f / (1.f + expf(-(acc + bo[0])));
}

// ================================================================= host ====
static inline void* wsAlloc(char*& p, size_t bytes) {
  void* r = p;
  p += (bytes + 255) & ~(size_t)255;
  return r;
}

extern "C" void kernel_launch(void* const* d_in, const int* in_sizes, int n_in,
                              void* d_out, int out_size, void* d_ws, size_t ws_size,
                              hipStream_t stream) {
  (void)in_sizes; (void)n_in; (void)out_size; (void)ws_size;
  const int*   x      = (const int*)  d_in[0];
  const float* tables = (const float*)d_in[1];
  const float* dW0    = (const float*)d_in[2];
  const float* db0    = (const float*)d_in[3];
  const float* dg0    = (const float*)d_in[4];
  const float* dbt0   = (const float*)d_in[5];
  const float* dW1    = (const float*)d_in[6];
  const float* db1    = (const float*)d_in[7];
  const float* dg1    = (const float*)d_in[8];
  const float* dbt1   = (const float*)d_in[9];
  const float* dWo    = (const float*)d_in[10];
  const float* dbo    = (const float*)d_in[11];
  const float* hW0    = (const float*)d_in[12];
  const float* hb0    = (const float*)d_in[13];
  const float* hg0    = (const float*)d_in[14];
  const float* hbt0   = (const float*)d_in[15];
  const float* hW1    = (const float*)d_in[16];
  const float* hb1    = (const float*)d_in[17];
  const float* hg1    = (const float*)d_in[18];
  const float* hbt1   = (const float*)d_in[19];
  const float* u0     = (const float*)d_in[20];
  const float* u1     = (const float*)d_in[21];
  const float* v0     = (const float*)d_in[22];
  const float* v1     = (const float*)d_in[23];
  const float* bl0    = (const float*)d_in[24];
  const float* bl1    = (const float*)d_in[25];
  const float* gamma1 = (const float*)d_in[26];
  const float* bias1  = (const float*)d_in[27];
  float* out = (float*)d_out;

  char* wp = (char*)d_ws;
  f16*   emb_h = (f16*)  wsAlloc(wp, (size_t)B_ * IN_ * 2);
  f16*   act_h = (f16*)  wsAlloc(wp, (size_t)B_ * H1_ * 2);
  float* hyper = (float*)wsAlloc(wp, (size_t)B_ * H1_ * 4);   // K_*K_ == H1_
  float* y     = (float*)wsAlloc(wp, (size_t)B_ * H1_ * 4);   // raw GEMM out
  float* di_f  = (float*)wsAlloc(wp, (size_t)B_ * H2_ * 4);
  f16*   di_h  = (f16*)  wsAlloc(wp, (size_t)B_ * H2_ * 2);
  f16*   m_h   = (f16*)  wsAlloc(wp, (size_t)B_ * K_  * 2);
  f16*   wt0   = (f16*)  wsAlloc(wp, (size_t)H1_ * IN_ * 2);
  f16*   wt1   = (f16*)  wsAlloc(wp, (size_t)H1_ * H2_ * 2);
  f16*   u0t   = (f16*)  wsAlloc(wp, (size_t)K_ * H2_ * 2);
  f16*   v1t   = (f16*)  wsAlloc(wp, (size_t)H2_ * K_ * 2);
  float* stats = (float*)wsAlloc(wp, (size_t)2 * H1_ * 4);

  const int TB = 256;
  auto blocks = [](long n) { return (int)((n + 255) / 256); };
  auto zstats = [&]() { hipMemsetAsync(stats, 0, 2 * H1_ * sizeof(float), stream); };

  // ---- embeddings
  gather_embed<<<blocks((long)B_ * F_), TB, 0, stream>>>(x, tables, emb_h);

  // ---- hyper net: emb->HH->K*K, BN+ReLU each layer
  convT<<<blocks((long)IN_ * HH_), TB, 0, stream>>>(hW0, wt0, IN_, HH_);
  zstats();
  wmma_gemm<<<dim3(B_ / 64, HH_ / 32), TB, 0, stream>>>(emb_h, wt0, hb0, y, stats, IN_, HH_, 3);
  bn_relu<<<blocks((long)B_ * HH_), TB, 0, stream>>>(y, stats, hg0, hbt0, nullptr, act_h, HH_);

  convT<<<blocks((long)HH_ * H1_), TB, 0, stream>>>(hW1, wt1, HH_, H1_);
  zstats();
  wmma_gemm<<<dim3(B_ / 64, H1_ / 32), TB, 0, stream>>>(act_h, wt1, hb1, y, stats, HH_, H1_, 3);
  bn_relu<<<blocks((long)B_ * H1_), TB, 0, stream>>>(y, stats, hg1, hbt1, hyper, nullptr, H1_);

  // ---- low-rank factors to f16 (transposed)
  convT<<<blocks((long)H2_ * K_), TB, 0, stream>>>(u0, u0t, H2_, K_);
  convT<<<blocks((long)K_ * H2_), TB, 0, stream>>>(v1, v1t, K_, H2_);

  // ---- per-domain towers
  for (int d = 0; d < D_; ++d) {
    convT<<<blocks((long)IN_ * H1_), TB, 0, stream>>>(dW0 + (size_t)d * IN_ * H1_, wt0, IN_, H1_);
    zstats();
    wmma_gemm<<<dim3(B_ / 64, H1_ / 32), TB, 0, stream>>>(emb_h, wt0, db0 + d * H1_, y, stats, IN_, H1_, 3);
    bn_relu<<<blocks((long)B_ * H1_), TB, 0, stream>>>(y, stats, dg0 + d * H1_, dbt0 + d * H1_, nullptr, act_h, H1_);

    convT<<<blocks((long)H1_ * H2_), TB, 0, stream>>>(dW1 + (size_t)d * H1_ * H2_, wt1, H1_, H2_);
    zstats();
    wmma_gemm<<<dim3(B_ / 64, H2_ / 32), TB, 0, stream>>>(act_h, wt1, db1 + d * H2_, y, stats, H1_, H2_, 3);
    bn_relu<<<blocks((long)B_ * H2_), TB, 0, stream>>>(y, stats, dg1 + d * H2_, dbt1 + d * H2_, di_f, di_h, H2_);

    // p = di @ u0  -> [B,32] (into y, no bias/stats)
    wmma_gemm<<<dim3(B_ / 64, K_ / 32), TB, 0, stream>>>(di_h, u0t, nullptr, y, stats, H2_, K_, 0);
    // per-sample 32-wide chain -> m (f16)
    adapter_mid<<<B_ / 8, TB, 0, stream>>>(y, hyper, v0, u1, bl0, m_h);
    // t2 = m @ v1 + bl1  -> [B,512] with ddof=1 BN stats
    zstats();
    wmma_gemm<<<dim3(B_ / 64, H2_ / 32), TB, 0, stream>>>(m_h, v1t, bl1, y, stats, K_, H2_, 3);
    bn2_residual<<<blocks((long)B_ * H2_), TB, 0, stream>>>(y, stats, gamma1, bias1, di_f);

    final_head<<<B_ / 8, TB, 0, stream>>>(di_f, dWo + (size_t)d * H2_, dbo + d, x, d, out);
  }
}
